// HeavyLayer_81020263071912
// MI455X (gfx1250) — matrix operand order repaired and checked by reading, not compiled, not run on hardware
//
#include <hip/hip_runtime.h>
#include <hip/hip_bf16.h>
#include <math.h>

typedef __attribute__((ext_vector_type(16))) __bf16 v16bf;
typedef __attribute__((ext_vector_type(8)))  __bf16 v8bf;
typedef __attribute__((ext_vector_type(8)))  float  v8f;
typedef __attribute__((ext_vector_type(4)))  int    v4i;

#define DM     1024   // D_MODEL
#define TOKENS 8192   // 4 * 2048
#define TK     32     // K-step per stage (one WMMA K)

// ---- CDNA5 async global->LDS copy (bypasses VGPRs, tracked by ASYNCcnt) ----
#if defined(__has_builtin)
#if __has_builtin(__builtin_amdgcn_global_load_async_to_lds_b128)
#define HAVE_ASYNC_LDS 1
#endif
#endif
#ifndef HAVE_ASYNC_LDS
#define HAVE_ASYNC_LDS 0
#endif

#if HAVE_ASYNC_LDS
typedef __attribute__((address_space(1))) v4i* gv4i_p;   // global v4i*
typedef __attribute__((address_space(3))) v4i* lv4i_p;   // LDS v4i*
#if __has_builtin(__builtin_amdgcn_s_wait_asynccnt)
#define WAIT_ASYNC(n) __builtin_amdgcn_s_wait_asynccnt(n)
#else
#define WAIT_ASYNC(n) asm volatile("s_wait_asynccnt %0" ::"i"(n) : "memory")
#endif
#define NBUF 3
#else
#define NBUF 2
#endif

// ---------------- f32 -> bf16 convert (grid-stride) ----------------
__global__ void cvt_f32_bf16(const float* __restrict__ in,
                             __bf16* __restrict__ out, int n) {
  int i = blockIdx.x * blockDim.x + threadIdx.x;
  int stride = gridDim.x * blockDim.x;
  for (; i < n; i += stride) out[i] = (__bf16)in[i];
}

// ------- transpose 1024x1024 f32 -> bf16 : Wt[n][k] = W[k][n] -------
__global__ void transpose_w(const float* __restrict__ W,
                            __bf16* __restrict__ Wt) {
  __shared__ __bf16 tile[32][33];
  const int n0 = blockIdx.x * 32;
  const int k0 = blockIdx.y * 32;
  const int tx = threadIdx.x;   // 0..31
  const int ty = threadIdx.y;   // 0..7
#pragma unroll
  for (int j = 0; j < 32; j += 8)
    tile[ty + j][tx] = (__bf16)W[(size_t)(k0 + ty + j) * DM + (n0 + tx)];
  __syncthreads();
#pragma unroll
  for (int j = 0; j < 32; j += 8)
    Wt[(size_t)(n0 + ty + j) * DM + (k0 + tx)] = tile[tx][ty + j];
}

static __device__ __forceinline__ v8bf lo8(v16bf v) {
  return __builtin_shufflevector(v, v, 0, 1, 2, 3, 4, 5, 6, 7);
}
static __device__ __forceinline__ v8bf hi8(v16bf v) {
  return __builtin_shufflevector(v, v, 8, 9, 10, 11, 12, 13, 14, 15);
}
static __device__ __forceinline__ v16bf cat8(v8bf a, v8bf b) {
  return __builtin_shufflevector(a, b, 0, 1, 2, 3, 4, 5, 6, 7,
                                 8, 9, 10, 11, 12, 13, 14, 15);
}

// ---------------- bf16 WMMA GEMM: C = A[M,K] * Bt[N,K]^T ----------------
// A row-major [M,1024] bf16; Bt[n][k] = W[k][n] bf16 (K-contiguous per column).
// Workgroup = 8 waves as 4(M) x 2(N); wave tile = 32x64:
//   2 A-fragments x 4 B-fragments -> 8 WMMA per K-step, each B fragment
//   feeds 2 WMMAs => 1 ds_load_b128 per WMMA (LDS bandwidth roofline).
// B panel (128 cols x 32 K) staged in LDS in fragment order; async triple
// buffer with prefetch distance 2, ASYNCcnt-throttled (fallback: ds_store).
// TANH==0: store bf16 to Ob.  TANH==1: store f32 tanh(acc+bias) to Of.
template <int TANH>
__global__ __launch_bounds__(256) void gemm_bf16_wmma(
    const __bf16* __restrict__ A,
    const __bf16* __restrict__ Bt,
    const float*  __restrict__ bias,
    __bf16* __restrict__ Ob,
    float*  __restrict__ Of) {
  __shared__ __align__(16) __bf16 ldsB[NBUF][2][256 * 8];  // NBUF * 8 KB

  const int tid  = threadIdx.x;
  const int wave = tid >> 5;
  const int lane = tid & 31;
  const int r16  = lane & 15;   // row (A) / column (B,C) within 16-tile
  const int hi   = lane >> 4;   // half-wave selector

  const int mrow  = wave >> 1;  // 0..3 : which 32-row band
  const int nhalf = wave & 1;   // 0..1 : which 64-col half

  const int m0 = blockIdx.x * 128 + mrow * 32;
  const int n0 = blockIdx.y * 128;
  const int nb = n0 + nhalf * 64;

  // Staging: slot i = f*32 + l holds what lane l of global fragment f needs:
  // Bt[n0 + f*16 + (l&15)][k0 + (l>>4)*16 .. +15]  (32 bytes).
  const int scol  = (tid >> 5) * 16 + (tid & 15);
  const int skoff = ((tid >> 4) & 1) * 16;
  const __bf16* Bsrc = Bt + (size_t)(n0 + scol) * DM + skoff;

  // A-fragment lane base: row m0+r16 (+16 for 2nd frag); contiguous K layout (k = 16*hi + s).
  const __bf16* Arow = A + (size_t)(m0 + r16) * DM + hi * 16;

  v8f acc[2][4] = {};

#if HAVE_ASYNC_LDS
  auto stage = [&](int k0, int buf) {
    __builtin_amdgcn_global_load_async_to_lds_b128(
        (gv4i_p)(const_cast<__bf16*>(Bsrc + k0)),
        (lv4i_p)&ldsB[buf][0][tid * 8], 0, 0);
    __builtin_amdgcn_global_load_async_to_lds_b128(
        (gv4i_p)(const_cast<__bf16*>(Bsrc + k0 + 8)),
        (lv4i_p)&ldsB[buf][1][tid * 8], 0, 0);
  };
  stage(0, 0);
  stage(TK, 1);
#else
  {
    v16bf s0 = *(const v16bf*)(Bsrc);
    *(v8bf*)&ldsB[0][0][tid * 8] = lo8(s0);
    *(v8bf*)&ldsB[0][1][tid * 8] = hi8(s0);
  }
#endif

  v8bf a00 = *(const v8bf*)(Arow);
  v8bf a01 = *(const v8bf*)(Arow + 8);
  v8bf a10 = *(const v8bf*)(Arow + 16 * DM);
  v8bf a11 = *(const v8bf*)(Arow + 16 * DM + 8);

#if HAVE_ASYNC_LDS
  WAIT_ASYNC(2);          // stage 0 resident (async loads complete in order)
#endif
  __syncthreads();

  int buf = 0;
  for (int k0 = 0; k0 < DM; k0 += TK) {
    const bool more = (k0 + TK) < DM;

#if HAVE_ASYNC_LDS
    if (k0 + 2 * TK < DM) stage(k0 + 2 * TK, (buf + 2) % 3);
#else
    v16bf snext;
    if (more) snext = *(const v16bf*)(Bsrc + k0 + TK);
#endif
    v8bf a00n, a01n, a10n, a11n;
    if (more) {
      const __bf16* an = Arow + k0 + TK;
      a00n = *(const v8bf*)(an);
      a01n = *(const v8bf*)(an + 8);
      a10n = *(const v8bf*)(an + 16 * DM);
      a11n = *(const v8bf*)(an + 16 * DM + 8);
    }

    const v16bf a0 = cat8(a00, a01);
    const v16bf a1 = cat8(a10, a11);
#pragma unroll
    for (int t = 0; t < 4; ++t) {
      const int slot = ((nhalf * 4 + t) * 32 + lane) * 8;
      v8bf blo = *(const v8bf*)&ldsB[buf][0][slot];
      v8bf bhi = *(const v8bf*)&ldsB[buf][1][slot];
      const v16bf b = cat8(blo, bhi);
      acc[0][t] = __builtin_amdgcn_wmma_f32_16x16x32_bf16(
          false, a0, false, b, (short)0, acc[0][t], false, false);
      acc[1][t] = __builtin_amdgcn_wmma_f32_16x16x32_bf16(
          false, a1, false, b, (short)0, acc[1][t], false, false);
    }

#if HAVE_ASYNC_LDS
    if (k0 + 2 * TK < DM) WAIT_ASYNC(2);       // stage k0+TK resident
    else if (more)        WAIT_ASYNC(0);
    buf = (buf == 2) ? 0 : buf + 1;
#else
    if (more) {
      const int nxt = buf ^ 1;
      *(v8bf*)&ldsB[nxt][0][tid * 8] = lo8(snext);
      *(v8bf*)&ldsB[nxt][1][tid * 8] = hi8(snext);
    }
    buf ^= 1;
#endif
    if (more) { a00 = a00n; a01 = a01n; a10 = a10n; a11 = a11n; }
    __syncthreads();
  }

  // Epilogue per ISA C/D layout: VGPR r -> M = +r + hi*8, N = col
#pragma unroll
  for (int am = 0; am < 2; ++am) {
#pragma unroll
    for (int t = 0; t < 4; ++t) {
      const int col = nb + t * 16 + r16;
#pragma unroll
      for (int r = 0; r < 8; ++r) {
        const int m = m0 + am * 16 + hi * 8 + r;
        const float v = acc[am][t][r];
        if (TANH) Of[(size_t)m * DM + col] = tanhf(v + bias[col]);
        else      Ob[(size_t)m * DM + col] = (__bf16)v;
      }
    }
  }
}

extern "C" void kernel_launch(void* const* d_in, const int* in_sizes, int n_in,
                              void* d_out, int out_size, void* d_ws,
                              size_t ws_size, hipStream_t stream) {
  const float* x    = (const float*)d_in[0];  // [4,2048,1024]
  const float* w    = (const float*)d_in[1];  // [1024,1024]
  const float* a1   = (const float*)d_in[2];  // [1024,1024]
  const float* a2   = (const float*)d_in[3];  // [1024,1024]
  const float* bias = (const float*)d_in[4];  // [1024]
  float* out = (float*)d_out;

  // Workspace layout (38 MB): [0,16M) xb (reused as h2), [16M,32M) h1,
  // [32M..38M) three transposed bf16 weights.
  char* ws = (char*)d_ws;
  __bf16* xb  = (__bf16*)(ws);
  __bf16* h1  = (__bf16*)(ws + ((size_t)16 << 20));
  __bf16* w1t = (__bf16*)(ws + ((size_t)32 << 20));
  __bf16* w2t = (__bf16*)(ws + ((size_t)34 << 20));
  __bf16* w3t = (__bf16*)(ws + ((size_t)36 << 20));
  __bf16* h2  = xb;  // x no longer needed after GEMM1

  // Prep: convert activations, transpose+convert weights.
  cvt_f32_bf16<<<4096, 256, 0, stream>>>(x, xb, TOKENS * DM);
  dim3 tb(32, 8), tg(DM / 32, DM / 32);
  transpose_w<<<tg, tb, 0, stream>>>(w,  w1t);
  transpose_w<<<tg, tb, 0, stream>>>(a1, w2t);
  transpose_w<<<tg, tb, 0, stream>>>(a2, w3t);

  // Chained GEMMs through the WMMA pipes.
  dim3 gg(TOKENS / 128, DM / 128);
  gemm_bf16_wmma<0><<<gg, 256, 0, stream>>>(xb, w1t, nullptr, h1, nullptr);
  gemm_bf16_wmma<0><<<gg, 256, 0, stream>>>(h1, w2t, nullptr, h2, nullptr);
  gemm_bf16_wmma<1><<<gg, 256, 0, stream>>>(h2, w3t, bias, nullptr, out);
}
